// OPTAttentionWithExtras_9053791060218
// MI455X (gfx1250) — matrix operand order repaired and checked by reading, not compiled, not run on hardware
//
#include <hip/hip_runtime.h>
#include <hip/hip_bf16.h>

// ---------------- problem dims (fixed by reference) ----------------
constexpr int B_ = 4, T_ = 1024, E_ = 2048, H_ = 32, D_ = 64;
constexpr int M_ROWS = B_ * T_;                 // 4096
constexpr float SCALING  = 0.125f;              // D^-0.5
constexpr float GAMMA_C  = -12.0f / 1024.0f;    // -ALPHA/MAX_SEQ
constexpr float ETA_C    = 1.0f;
constexpr float NEG_MIN  = -3.402823466e38f;

typedef __attribute__((ext_vector_type(16))) _Float16     v16h;
typedef __attribute__((ext_vector_type(8)))  _Float16     v8h;
typedef __attribute__((ext_vector_type(8)))  float        v8f;
typedef __attribute__((ext_vector_type(4)))  float        f32x4;
typedef __attribute__((ext_vector_type(4)))  unsigned int u32x4;
typedef __attribute__((ext_vector_type(8)))  int          i32x8;
typedef __attribute__((ext_vector_type(4)))  int          i32x4;

#if defined(__has_builtin)
#  if __has_builtin(__builtin_amdgcn_tensor_load_to_lds) && \
      __has_builtin(__builtin_amdgcn_s_wait_tensorcnt)
#    define HAVE_TDM 1
#  endif
#endif
#ifndef HAVE_TDM
#  define HAVE_TDM 0
#endif

__device__ __forceinline__ v8f wmma_f16(v16h a, v16h b, v8f c) {
  // D = A(16x32 f16) * B(32x16 f16) + C(16x16 f32)
  return __builtin_amdgcn_wmma_f32_16x16x32_f16(false, a, false, b, (short)0, c,
                                                false, false);
}

// ---- fragment loaders (per ISA 7.12.2 VGPR layouts, wave32) ----
// A-frag 16x32 f16: lane holds row M=lane&15; K sub-ranges {kb..kb+7, kb+16..kb+23}
// with kb = (lane>>4)*8 folded into the kb argument by the caller.
__device__ __forceinline__ v16h cvt_afrag_f32(const float* __restrict__ p, int kb) {
  f32x4 f0 = *(const f32x4*)(p + kb);
  f32x4 f1 = *(const f32x4*)(p + kb + 4);
  f32x4 f2 = *(const f32x4*)(p + kb + 16);
  f32x4 f3 = *(const f32x4*)(p + kb + 20);
  v16h h;
#pragma unroll
  for (int j = 0; j < 4; ++j) {
    h[j]      = (_Float16)f0[j];
    h[4 + j]  = (_Float16)f1[j];
    h[8 + j]  = (_Float16)f2[j];
    h[12 + j] = (_Float16)f3[j];
  }
  return h;
}
// B-frag 32x16 f16: lane holds col N=lane&15; K sub-ranges {kb..kb+7, kb+16..kb+23}, kb=(lane>>4)*8.
__device__ __forceinline__ v16h cvt_bfrag_f32(const float* __restrict__ p, int kb) {
  f32x4 f0 = *(const f32x4*)(p + kb);
  f32x4 f1 = *(const f32x4*)(p + kb + 4);
  f32x4 f2 = *(const f32x4*)(p + kb + 16);
  f32x4 f3 = *(const f32x4*)(p + kb + 20);
  v16h h;
#pragma unroll
  for (int j = 0; j < 4; ++j) {
    h[j]      = (_Float16)f0[j];
    h[4 + j]  = (_Float16)f1[j];
    h[8 + j]  = (_Float16)f2[j];
    h[12 + j] = (_Float16)f3[j];
  }
  return h;
}
__device__ __forceinline__ v16h ld_afrag_h(const _Float16* __restrict__ p, int kb) {
  v8h lo = *(const v8h*)(p + kb);
  v8h hi = *(const v8h*)(p + kb + 16);
  v16h h;
#pragma unroll
  for (int j = 0; j < 8; ++j) { h[j] = lo[j]; h[8 + j] = hi[j]; }
  return h;
}
__device__ __forceinline__ v16h ld_bfrag_h(const _Float16* __restrict__ p, int kb) {
  v8h lo = *(const v8h*)(p + kb);
  v8h hi = *(const v8h*)(p + kb + 16);
  v16h h;
#pragma unroll
  for (int j = 0; j < 8; ++j) { h[j] = lo[j]; h[8 + j] = hi[j]; }
  return h;
}
// overloaded A loader (float = convert, f16 = direct) for the templated GEMM
__device__ __forceinline__ v16h ld_afrag(const float* __restrict__ p, int kb)    { return cvt_afrag_f32(p, kb); }
__device__ __forceinline__ v16h ld_afrag(const _Float16* __restrict__ p, int kb) { return ld_afrag_h(p, kb); }

// ================= GEMM: C[M,N] = (A[M,K] @ W[N,K]^T + bias) * scale =========
// Grid: (M/64, N/256); 256 threads = 8 waves in a 2(m) x 4(n) grid;
// each wave owns a 32x64 C tile = 2x4 WMMA accumulators (8 WMMA per K-step).
// A tile (64 rows x 32 K) is staged in LDS via the Tensor Data Mover, double
// buffered: TENSOR_LOAD_TO_LDS(next) || compute(cur) ; s_wait_tensorcnt; barrier.
// OUT_MODE: 0 = f32 row-major, 1 = f16 row-major, 2 = f16 head-transposed [B,H,D,T]
template <typename AT, int OUT_MODE>
__global__ __launch_bounds__(256) void wmma_gemm_xwt(
    const AT* __restrict__ A, const float* __restrict__ W,
    const float* __restrict__ bias, void* __restrict__ Cout,
    int K, int N, float scale)
{
  __shared__ AT stageA[2][64 * 32];   // 16 KB (f32) / 8 KB (f16)

  const int lane = threadIdx.x & 31;
  const int wave = threadIdx.x >> 5;
  const int wm   = wave & 1;
  const int wn   = wave >> 1;
  const int m0   = blockIdx.x * 64  + wm * 32;
  const int n0   = blockIdx.y * 256 + wn * 64;
  const int lrow = lane & 15;
  const int ah   = (lane >> 4) * 8;    // A-frag K sub-base
  const int bh16 = (lane >> 4) * 8;    // B-frag K sub-base

  const AT* atile = A + (size_t)(blockIdx.x * 64) * K;  // WG's 64 A rows
  const float* wr[4];
#pragma unroll
  for (int j = 0; j < 4; ++j) wr[j] = W + (size_t)(n0 + j * 16 + lrow) * K;

  // ---- async A-tile staging (TDM descriptor per ISA ch.8) ----
  auto stage_issue = [&](int buf, int kk) {
#if HAVE_TDM
    if (wave == 0) {
      const unsigned long long ga = (unsigned long long)(size_t)(atile + kk);
      const unsigned lds_off = (unsigned)(size_t)(void*)&stageA[buf][0];
      u32x4 g0;
      g0[0] = 1u;                                        // count=1 (valid user D#)
      g0[1] = lds_off;                                   // LDS byte address
      g0[2] = (unsigned)(ga & 0xffffffffu);              // global_addr[31:0]
      g0[3] = (unsigned)((ga >> 32) & 0x01ffffffu) | (2u << 30);  // [56:32] | type=2
      constexpr unsigned dsz = (sizeof(AT) == 4) ? 2u : 1u;       // 4B : 2B
      i32x8 g1;
      g1[0] = (int)(dsz << 16);        // wg_mask=0 (not in cluster), data_size
      g1[1] = (int)(32u << 16);        // tensor_dim0[15:0]  = 32 (tile-local)
      g1[2] = (int)(64u << 16);        // tensor_dim1[15:0]  = 64
      g1[3] = (int)(32u << 16);        // tile_dim0 = 32
      g1[4] = (int)64u;                // tile_dim1 = 64, tile_dim2 = 0
      g1[5] = (int)(unsigned)K;        // tensor_dim0_stride[31:0] = K elements
      g1[6] = 0;                       // stride hi / dim1_stride lo
      g1[7] = 0;
      const i32x4 gz = {0, 0, 0, 0};
#if __clang_major__ >= 23
      const i32x8 gz8 = {0, 0, 0, 0, 0, 0, 0, 0};
      __builtin_amdgcn_tensor_load_to_lds(g0, g1, gz, gz, gz8, 0);
#else
      __builtin_amdgcn_tensor_load_to_lds(g0, g1, gz, gz, 0);
#endif
    }
#else
    // cooperative fallback copy (256 threads, 64x32 elements)
    for (int idx = threadIdx.x; idx < 64 * 32; idx += 256) {
      const int r = idx >> 5, c = idx & 31;
      stageA[buf][idx] = atile[(size_t)r * K + kk + c];
    }
#endif
  };
  auto stage_wait = [&]() {
    asm volatile("s_wait_dscnt 0" ::: "memory");   // close read-vs-DMA WAR window
#if HAVE_TDM
    if (wave == 0) __builtin_amdgcn_s_wait_tensorcnt(0);
#endif
    __syncthreads();
  };

  v8f acc[2][4] = {};

  stage_issue(0, 0);
  stage_wait();
  int cur = 0;

  for (int kk = 0; kk < K; kk += 32) {
    const bool more = (kk + 32 < K);
    if (more) stage_issue(cur ^ 1, kk + 32);       // async fill of next slab
    if (more) {                                    // warm L2 for next W slab
      __builtin_prefetch(wr[0] + kk + 32, 0, 3);
      __builtin_prefetch(wr[2] + kk + 32, 0, 3);
    }

    const AT* as = &stageA[cur][0];
    v16h af0 = ld_afrag(as + (size_t)(wm * 32 + lrow) * 32, ah);
    v16h af1 = ld_afrag(as + (size_t)(wm * 32 + 16 + lrow) * 32, ah);
#pragma unroll
    for (int j = 0; j < 4; ++j) {
      v16h bf = cvt_bfrag_f32(wr[j], kk + bh16);
      acc[0][j] = wmma_f16(af0, bf, acc[0][j]);
      acc[1][j] = wmma_f16(af1, bf, acc[1][j]);
    }

    if (more) { stage_wait(); cur ^= 1; }
  }

  // C layout: lane l, VGPR r -> M = r + 8*(l>>4), N = l&15
#pragma unroll
  for (int i = 0; i < 2; ++i)
#pragma unroll
    for (int j = 0; j < 4; ++j) {
      const int   col  = n0 + j * 16 + lrow;
      const float bcol = bias[col];
#pragma unroll
      for (int r = 0; r < 8; ++r) {
        const int   row  = m0 + i * 16 + r + ((lane >> 4) << 3);
        const float vout = (acc[i][j][r] + bcol) * scale;
        if constexpr (OUT_MODE == 0) {
          ((float*)Cout)[(size_t)row * N + col] = vout;
        } else if constexpr (OUT_MODE == 1) {
          ((_Float16*)Cout)[(size_t)row * N + col] = (_Float16)vout;
        } else {
          // head-transposed: [b, h, d, t] with h=col/64, d=col%64, b=row/1024, t=row%1024
          const int hh = col >> 6, dd = col & (D_ - 1);
          const int bb = row >> 10, tt = row & (T_ - 1);
          ((_Float16*)Cout)[(((size_t)bb * H_ + hh) * D_ + dd) * T_ + tt] = (_Float16)vout;
        }
      }
    }
}

// ============== Attention core: clipped softmax + gated PV ===================
__device__ __forceinline__ v8f qk_tile(const _Float16* __restrict__ kbase, int st,
                                       int lrow, int bh16,
                                       const v16h& aq0, const v16h& aq1) {
  const _Float16* kr = kbase + (size_t)(st * 16 + lrow) * E_;
  v16h b0 = ld_bfrag_h(kr, bh16);        // d 0..31
  v16h b1 = ld_bfrag_h(kr, 32 + bh16);   // d 32..63
  v8f s = {};
  s = wmma_f16(aq0, b0, s);
  s = wmma_f16(aq1, b1, s);
  return s;
}

// Grid: (B*H, T/128); 256 threads = 8 waves; each wave handles 16 query rows.
// q,k: f16 [B*T, E] row-major.  vT: f16 [B,H,D,T].  out: f16 [B*T, E].
__global__ __launch_bounds__(256) void attn_kernel(
    const _Float16* __restrict__ q, const _Float16* __restrict__ k,
    const _Float16* __restrict__ vT, const float* __restrict__ x,
    const float* __restrict__ gate_w, const float* __restrict__ gate_b,
    _Float16* __restrict__ out)
{
  __shared__ _Float16 lds_p[8][16 * 32];   // per-wave prob tile re-layout
  __shared__ float    lds_g[8][16];        // per-wave row gates

  const int lane = threadIdx.x & 31;
  const int wave = threadIdx.x >> 5;
  const int lrow = lane & 15;
  const int ah   = (lane >> 4) * 8;
  const int bh16 = (lane >> 4) * 8;

  const int bh_idx = blockIdx.x;
  const int b = bh_idx >> 5;           // H_ == 32
  const int h = bh_idx & (H_ - 1);
  const int m0 = blockIdx.y * 128 + wave * 16;

  const size_t base_t = (size_t)b * T_;
  const _Float16* qb = q + (base_t + m0) * E_ + h * D_;
  const _Float16* kb = k + base_t * E_ + h * D_;
  const _Float16* vh = vT + ((size_t)(b * H_ + h) * D_) * T_;

  // Q A-fragments (rows m0..m0+15, d 0..63), loaded once
  const _Float16* qr = qb + (size_t)lrow * E_;
  v16h aq0 = ld_afrag_h(qr, ah);
  v16h aq1 = ld_afrag_h(qr, 32 + ah);

  // per-row sigmoid gate: sigmoid(x_heads . gate_w[h] + gate_b[h])
  if (lane < 16) {
    const float* xr = x + (base_t + m0 + lane) * E_ + h * D_;
    const float* gw = gate_w + h * D_;
    float a = gate_b[h];
#pragma unroll
    for (int d = 0; d < D_; ++d) a = fmaf(xr[d], gw[d], a);
    lds_g[wave][lane] = 1.0f / (1.0f + __expf(-a));
  }

  const int ntiles = (m0 >> 4) + 1;  // causal: key tiles with s0 <= m0+15

  // ---- pass 1: per-lane deferred (max, exp-sum); no per-tile shuffles ----
  float lm[8], ls[8];
#pragma unroll
  for (int r = 0; r < 8; ++r) { lm[r] = NEG_MIN; ls[r] = 0.0f; }

  for (int st = 0; st < ntiles; ++st) {
    v8f s = qk_tile(kb, st, lrow, bh16, aq0, aq1);
    const int colg = st * 16 + lrow;
#pragma unroll
    for (int r = 0; r < 8; ++r) {
      const int rowg = m0 + r + ((lane >> 4) << 3);
      const float sc = (colg <= rowg) ? s[r] : NEG_MIN;
      const float mn = fmaxf(lm[r], sc);
      ls[r] = ls[r] * __expf(lm[r] - mn) + __expf(sc - mn);
      lm[r] = mn;
    }
  }
  // single cross-lane combine (16-lane halves hold 16 columns of each row)
  float rm[8], invl[8];
#pragma unroll
  for (int r = 0; r < 8; ++r) {
    float m = lm[r], l = ls[r];
#pragma unroll
    for (int off = 1; off < 16; off <<= 1) {
      const float mo = __shfl_xor(m, off);
      const float lo = __shfl_xor(l, off);
      const float mn = fmaxf(m, mo);
      l = l * __expf(m - mn) + lo * __expf(mo - mn);
      m = mn;
    }
    rm[r] = m;
    invl[r] = 1.0f / l;
  }

  // ---- pass 2: recompute scores, clipped softmax, P·V via WMMA ----
  v8f oacc[4] = {};
  _Float16* lp = &lds_p[wave][0];
  const int npairs = (ntiles + 1) >> 1;
  for (int sp = 0; sp < npairs; ++sp) {
#pragma unroll
    for (int hf = 0; hf < 2; ++hf) {
      const int st = sp * 2 + hf;
      if (st < ntiles) {
        v8f s = qk_tile(kb, st, lrow, bh16, aq0, aq1);
        const int colg = st * 16 + lrow;
#pragma unroll
        for (int r = 0; r < 8; ++r) {
          const int rowg = m0 + r + ((lane >> 4) << 3);
          const float sc = (colg <= rowg) ? s[r] : NEG_MIN;
          const float p  = __expf(sc - rm[r]) * invl[r];
          const float cp = fminf(fmaxf((ETA_C - GAMMA_C) * p + GAMMA_C, 0.0f), 1.0f);
          lp[(r + ((lane >> 4) << 3)) * 32 + hf * 16 + lrow] = (_Float16)cp;
        }
      } else {
#pragma unroll
        for (int r = 0; r < 8; ++r)
          lp[(r + ((lane >> 4) << 3)) * 32 + hf * 16 + lrow] = (_Float16)0.0f;
      }
    }
    // LDS store -> load relayout (C tile layout -> A fragment layout)
    asm volatile("s_wait_dscnt 0" ::: "memory");
    v16h ap = ld_afrag_h(lp + lrow * 32, ah);

#pragma unroll
    for (int j = 0; j < 4; ++j) {            // 4 d-tiles of 16 cols
      // vT row for col d = j*16+lrow; contiguous K (= key index) loads
      const _Float16* vr = vh + (size_t)(j * 16 + lrow) * T_;
      v16h bvf = ld_bfrag_h(vr, sp * 32 + bh16);
      oacc[j] = wmma_f16(ap, bvf, oacc[j]);
    }
  }

  // ---- epilogue: gate and store f16 to [B,T, h*D+d] (i.e. [B*T, E]) ----
  asm volatile("s_wait_dscnt 0" ::: "memory");
  _Float16* ob = out + (base_t + m0) * E_ + h * D_;
#pragma unroll
  for (int r = 0; r < 8; ++r) {
    const int   mr = r + ((lane >> 4) << 3);
    const float g  = lds_g[wave][mr];
#pragma unroll
    for (int j = 0; j < 4; ++j)
      ob[(size_t)mr * E_ + j * 16 + lrow] = (_Float16)(oacc[j][r] * g);
  }
}

// =============================== launcher ====================================
extern "C" void kernel_launch(void* const* d_in, const int* in_sizes, int n_in,
                              void* d_out, int out_size, void* d_ws, size_t ws_size,
                              hipStream_t stream) {
  const float* x  = (const float*)d_in[0];
  // d_in[1] = attention_mask (causal; implemented analytically)
  const float* Wq = (const float*)d_in[2];
  const float* bq = (const float*)d_in[3];
  const float* Wk = (const float*)d_in[4];
  const float* bk = (const float*)d_in[5];
  const float* Wv = (const float*)d_in[6];
  const float* bv = (const float*)d_in[7];
  const float* Wo = (const float*)d_in[8];
  const float* bo = (const float*)d_in[9];
  const float* gw = (const float*)d_in[10];
  const float* gb = (const float*)d_in[11];

  const size_t MN = (size_t)M_ROWS * E_;
  _Float16* qh    = (_Float16*)d_ws;         // 16 MB, [B*T, E] f16
  _Float16* kh    = qh + MN;                 // 16 MB, [B*T, E] f16
  _Float16* vt    = kh + MN;                 // 16 MB, [B,H,D,T] f16
  _Float16* attnh = vt + MN;                 // 16 MB, [B*T, E] f16

  dim3 blk(256);
  dim3 gg(M_ROWS / 64, E_ / 256);            // 64 x 8 workgroups

  wmma_gemm_xwt<float, 1><<<gg, blk, 0, stream>>>(x, Wq, bq, (void*)qh, E_, E_, SCALING);
  wmma_gemm_xwt<float, 1><<<gg, blk, 0, stream>>>(x, Wk, bk, (void*)kh, E_, E_, 1.0f);
  wmma_gemm_xwt<float, 2><<<gg, blk, 0, stream>>>(x, Wv, bv, (void*)vt, E_, E_, 1.0f);

  attn_kernel<<<dim3(B_ * H_, T_ / 128), blk, 0, stream>>>(qh, kh, vt, x, gw, gb, attnh);

  wmma_gemm_xwt<_Float16, 0><<<gg, blk, 0, stream>>>(attnh, Wo, bo, d_out, E_, E_, 1.0f);
}
